// Model_39676907883512
// MI455X (gfx1250) — matrix operand order repaired and checked by reading, not compiled, not run on hardware
//
#include <hip/hip_runtime.h>
#include <math.h>

typedef __attribute__((ext_vector_type(16))) _Float16 v16h;
typedef __attribute__((ext_vector_type(2)))  _Float16 h2;
typedef __attribute__((ext_vector_type(8)))  float    v8f;
typedef _Float16 half_t;

#define NB    64      // batch*heads
#define SEQ   2048
#define DIM   128
#define BC    32      // kv rows per tile iteration
#define BR_WG 128     // q rows per workgroup (8 waves x 16)
#define NWAVES 8

// D = A(16x32 f16) * B(32x16 f16) + C(16x16 f32), wave32 WMMA
static __device__ __forceinline__ v8f wmma_f16(v16h a, v16h b, v8f c) {
    return __builtin_amdgcn_wmma_f32_16x16x32_f16(false, a, false, b, (short)0, c,
                                                  false, false);
}

// k-offset inside a 32-wide K chunk for half index i of the 16-bit A-matrix
// (contiguous order, matching the B fragments): VGPR j -> K=2j,2j+1 ;
// lanes 16-31 add +16.  (i = 2*j + (i&1))
static __device__ __forceinline__ int a_kbase(int j, int lhi) {
    return 2 * j + (lhi << 4);
}

union HU { h2 h; int u; };

__global__ __launch_bounds__(256)
void fa_wmma_kernel(const float* __restrict__ Q,
                    const float* __restrict__ K,
                    const float* __restrict__ V,
                    const float* __restrict__ scale_factor,
                    float* __restrict__ Out) {
    __shared__ half_t lds_k[2][BC * DIM];          // K tile, row-major [kv][d]
    __shared__ half_t lds_vt[2][DIM * BC];         // V tile, transposed [d][kv]
    __shared__ half_t lds_p[NWAVES * 16 * BC];     // per-wave P scratch [m][n]

    const int tid  = threadIdx.x;
    const int wave = tid >> 5;
    const int lane = tid & 31;
    const int lhi  = lane >> 4;   // 0: lanes 0-15, 1: lanes 16-31
    const int llo  = lane & 15;

    const int b     = blockIdx.y;
    const int qrow0 = blockIdx.x * BR_WG + wave * 16;
    const size_t baseBSD = (size_t)b * SEQ * DIM;

    const float qscale = 1.0f / scale_factor[0];

    // ---- Load Q fragments once (A layout), pre-scaled by 1/sqrt(D) ----
    v16h qf[4];
    {
        const size_t rowoff = baseBSD + (size_t)(qrow0 + llo) * DIM;
#pragma unroll
        for (int c = 0; c < 4; ++c) {
#pragma unroll
            for (int j = 0; j < 8; ++j) {
                int d = 32 * c + a_kbase(j, lhi);
                qf[c][2 * j]     = (half_t)(Q[rowoff + d] * qscale);
                qf[c][2 * j + 1] = (half_t)(Q[rowoff + d + 1] * qscale);
            }
        }
    }

    // All-ones B fragment (layout-invariant) for row-sum matmul
    v16h onesB;
#pragma unroll
    for (int i = 0; i < 16; ++i) onesB[i] = (half_t)1.0f;

    // ---- Accumulator & online-softmax state ----
    v8f acc[8];
    const v8f vzero = {};
#pragma unroll
    for (int c = 0; c < 8; ++c) acc[c] = vzero;
    float mrow[8], lrow[8];
#pragma unroll
    for (int r = 0; r < 8; ++r) { mrow[r] = -1e30f; lrow[r] = 0.0f; }

    half_t* pScratch = &lds_p[wave * 16 * BC];
    const int ntiles = SEQ / BC;

    // ---- Prologue: stage tile 0 into buffer 0 ----
    float4 kreg[4], vreg[4];
#pragma unroll
    for (int it = 0; it < 4; ++it) {
        int blk = tid + it * 256;                  // float4 index (0..1023)
        kreg[it] = ((const float4*)(K + baseBSD))[blk];
        vreg[it] = ((const float4*)(V + baseBSD))[blk];
    }
#pragma unroll
    for (int it = 0; it < 4; ++it) {
        int blk = tid + it * 256;
        int h0  = blk * 4;
        lds_k[0][h0 + 0] = (half_t)kreg[it].x;
        lds_k[0][h0 + 1] = (half_t)kreg[it].y;
        lds_k[0][h0 + 2] = (half_t)kreg[it].z;
        lds_k[0][h0 + 3] = (half_t)kreg[it].w;
        int row = h0 >> 7, d0 = h0 & 127;
        lds_vt[0][(d0 + 0) * BC + row] = (half_t)vreg[it].x;
        lds_vt[0][(d0 + 1) * BC + row] = (half_t)vreg[it].y;
        lds_vt[0][(d0 + 2) * BC + row] = (half_t)vreg[it].z;
        lds_vt[0][(d0 + 3) * BC + row] = (half_t)vreg[it].w;
    }
    __syncthreads();

    int p = 0;
    for (int kt = 0; kt < ntiles; ++kt) {
        // ---- Issue next tile's global loads early (latency hides under WMMA) --
        const bool more = (kt + 1) < ntiles;
        if (more) {
            const size_t nb = baseBSD + (size_t)(kt + 1) * BC * DIM;
#pragma unroll
            for (int it = 0; it < 4; ++it) {
                int blk = tid + it * 256;
                kreg[it] = ((const float4*)(K + nb))[blk];
                vreg[it] = ((const float4*)(V + nb))[blk];
            }
        }

        const half_t* kbuf = lds_k[p];
        const half_t* vbuf = lds_vt[p];

        // ---- Preload all K^T B-fragments, then the QK^T WMMA chain ----
        v16h bk[8];
#pragma unroll
        for (int c = 0; c < 4; ++c) {
            bk[c]     = *(const v16h*)&kbuf[llo * DIM + 32 * c + 16 * lhi];
            bk[4 + c] = *(const v16h*)&kbuf[(16 + llo) * DIM + 32 * c + 16 * lhi];
        }
        v8f c0 = vzero, c1 = vzero;
#pragma unroll
        for (int c = 0; c < 4; ++c) {
            c0 = wmma_f16(qf[c], bk[c], c0);
            c1 = wmma_f16(qf[c], bk[4 + c], c1);
        }

        // ---- Row max: packed f16x2 butterfly in 16-lane halves (VALU + 16 shfl)
        h2 pm[4];
#pragma unroll
        for (int r2 = 0; r2 < 4; ++r2) {
            float a = fmaxf(c0[2 * r2],     c1[2 * r2]);
            float bq = fmaxf(c0[2 * r2 + 1], c1[2 * r2 + 1]);
            h2 t; t.x = (half_t)a; t.y = (half_t)bq;
            pm[r2] = t;
        }
#pragma unroll
        for (int mk = 1; mk <= 8; mk <<= 1) {
#pragma unroll
            for (int r2 = 0; r2 < 4; ++r2) {
                HU cur; cur.h = pm[r2];
                HU oth; oth.u = __shfl_xor(cur.u, mk);
                pm[r2] = __builtin_elementwise_max(cur.h, oth.h);
            }
        }

        // ---- exp, rescale, store P (f16) to per-wave scratch ----
        float alpha[8];
#pragma unroll
        for (int r = 0; r < 8; ++r) {
            h2 pmv = pm[r >> 1];
            float mx = (float)((r & 1) ? pmv.y : pmv.x);
            float mnew = fmaxf(mrow[r], mx);
            alpha[r]   = __expf(mrow[r] - mnew);
            float e0 = __expf(c0[r] - mnew);
            float e1 = __expf(c1[r] - mnew);
            mrow[r] = mnew;
            int m = r + (lhi << 3);
            pScratch[m * BC + llo]      = (half_t)e0;
            pScratch[m * BC + 16 + llo] = (half_t)e1;
#pragma unroll
            for (int c = 0; c < 8; ++c) acc[c][r] *= alpha[r];
        }

        // ---- Reload P in A-layout (same-wave LDS ordering; 8 x b32 loads) ----
        v16h pa;
#pragma unroll
        for (int j = 0; j < 8; ++j) {
            h2 t = *(const h2*)&pScratch[llo * BC + a_kbase(j, lhi)];
            pa[2 * j]     = t.x;
            pa[2 * j + 1] = t.y;
        }

        // ---- Row sums via WMMA against all-ones B; update l ----
        v8f rs = wmma_f16(pa, onesB, vzero);
#pragma unroll
        for (int r = 0; r < 8; ++r) lrow[r] = lrow[r] * alpha[r] + rs[r];

        // ---- O += P * V ----
#pragma unroll
        for (int c = 0; c < 8; ++c) {
            const v16h bv = *(const v16h*)&vbuf[(16 * c + llo) * BC + 16 * lhi];
            acc[c] = wmma_f16(pa, bv, acc[c]);
        }

        // ---- Convert & store next tile into the other buffer ----
        if (more) {
            half_t* kb = lds_k[p ^ 1];
            half_t* vb = lds_vt[p ^ 1];
#pragma unroll
            for (int it = 0; it < 4; ++it) {
                int blk = tid + it * 256;
                int h0  = blk * 4;
                kb[h0 + 0] = (half_t)kreg[it].x;
                kb[h0 + 1] = (half_t)kreg[it].y;
                kb[h0 + 2] = (half_t)kreg[it].z;
                kb[h0 + 3] = (half_t)kreg[it].w;
                int row = h0 >> 7, d0 = h0 & 127;
                vb[(d0 + 0) * BC + row] = (half_t)vreg[it].x;
                vb[(d0 + 1) * BC + row] = (half_t)vreg[it].y;
                vb[(d0 + 2) * BC + row] = (half_t)vreg[it].z;
                vb[(d0 + 3) * BC + row] = (half_t)vreg[it].w;
            }
        }
        __syncthreads();
        p ^= 1;
    }

    // ---- Epilogue: normalize by row sums and store fp32 ----
    float rinv[8];
#pragma unroll
    for (int r = 0; r < 8; ++r) rinv[r] = 1.0f / lrow[r];
#pragma unroll
    for (int c = 0; c < 8; ++c) {
#pragma unroll
        for (int r = 0; r < 8; ++r) {
            int row = qrow0 + r + (lhi << 3);
            int col = 16 * c + llo;
            Out[baseBSD + (size_t)row * DIM + col] = acc[c][r] * rinv[r];
        }
    }
}

extern "C" void kernel_launch(void* const* d_in, const int* in_sizes, int n_in,
                              void* d_out, int out_size, void* d_ws, size_t ws_size,
                              hipStream_t stream) {
    (void)in_sizes; (void)n_in; (void)d_ws; (void)ws_size; (void)out_size;
    const float* Q  = (const float*)d_in[0];
    const float* Kp = (const float*)d_in[1];
    const float* V  = (const float*)d_in[2];
    const float* sf = (const float*)d_in[3];
    float* Out = (float*)d_out;

    dim3 grid(SEQ / BR_WG, NB);
    dim3 block(256);
    fa_wmma_kernel<<<grid, block, 0, stream>>>(Q, Kp, V, sf, Out);
}
